// LinearCritic_59571196396061
// MI455X (gfx1250) — compile-verified, matrix operand order checked by reading, not run on hardware
//
#include <hip/hip_runtime.h>
#include <math.h>

// ---------------------------------------------------------------------------
// Problem constants (from reference)
// ---------------------------------------------------------------------------
#define NBATCH   4096
#define LATENT   2048
#define PROJ     128
#define M2       8192            // 2*NBATCH
#define BN_EPS   1e-5f
#define COS_EPS  1e-8f
#define INV_T    2.0f            // 1/TEMP

typedef __bf16 v16bf __attribute__((ext_vector_type(16)));
typedef float  v8f   __attribute__((ext_vector_type(8)));

// ---------------------------------------------------------------------------
// Helpers
// ---------------------------------------------------------------------------
__device__ __forceinline__ __bf16 f2bf(float f) {
    union { float f; unsigned u; } a; a.f = f;
    unsigned r = a.u + 0x7FFFu + ((a.u >> 16) & 1u);   // round-to-nearest-even
    union { unsigned short s; __bf16 b; } o;
    o.s = (unsigned short)(r >> 16);
    return o.b;
}

// Load one 16x32 bf16 A/B fragment for v_wmma_f32_16x16x32_bf16.
// Per ISA layout: element i of lane l holds K = (i/8)*16 + (l/16)*8 + (i%8),
// i.e. two contiguous 16-byte chunks per lane.
__device__ __forceinline__ v16bf load_frag(const __bf16* __restrict__ row, int k0) {
    const int half8 = ((threadIdx.x >> 4) & 1) * 8;
    union { v16bf v; float4 f[2]; } u;
    u.f[0] = *reinterpret_cast<const float4*>(row + k0 + half8);
    u.f[1] = *reinterpret_cast<const float4*>(row + k0 + 16 + half8);
    return u.v;
}

__device__ __forceinline__ v8f wmma_bf16(v16bf a, v16bf b, v8f c) {
    return __builtin_amdgcn_wmma_f32_16x16x32_bf16(
        /*neg_a=*/false, a, /*neg_b=*/false, b,
        /*c_mod=*/(short)0, c, /*reuse_a=*/false, /*reuse_b=*/false);
}

// ---------------------------------------------------------------------------
// Kernel 1: f32 -> bf16 conversion
// ---------------------------------------------------------------------------
__global__ void cvt_f32_bf16(const float* __restrict__ s, __bf16* __restrict__ d, int n) {
    int i = blockIdx.x * blockDim.x + threadIdx.x;
    if (i < n) d[i] = f2bf(s[i]);
}

// ---------------------------------------------------------------------------
// Kernel 2: GEMM1  X[8192,2048] = H[8192,2048] @ W1[2048,2048]^T   (NT, bf16->f32)
// Block = 256 threads = 8 waves; each wave owns a 64x64 tile (4x4 WMMA tiles).
// ---------------------------------------------------------------------------
__global__ __launch_bounds__(256) void gemm1_kernel(
        const __bf16* __restrict__ H, const __bf16* __restrict__ W,
        float* __restrict__ X) {
    const int K = LATENT, LDC = LATENT;
    const int lane = threadIdx.x & 31, wave = threadIdx.x >> 5;
    const int tm = blockIdx.y * 128 + (wave & 1) * 64;
    const int tn = blockIdx.x * 256 + (wave >> 1) * 64;
    const int r16 = lane & 15;

    const __bf16* ar[4]; const __bf16* br[4];
#pragma unroll
    for (int i = 0; i < 4; ++i) {
        ar[i] = H + (size_t)(tm + i * 16 + r16) * K;
        br[i] = W + (size_t)(tn + i * 16 + r16) * K;
    }
    v8f acc[4][4];
#pragma unroll
    for (int i = 0; i < 4; ++i)
#pragma unroll
        for (int j = 0; j < 4; ++j) acc[i][j] = (v8f)0.f;

    for (int k = 0; k < K; k += 32) {
        v16bf a[4], b[4];
#pragma unroll
        for (int i = 0; i < 4; ++i) a[i] = load_frag(ar[i], k);
#pragma unroll
        for (int j = 0; j < 4; ++j) b[j] = load_frag(br[j], k);
#pragma unroll
        for (int i = 0; i < 4; ++i)
#pragma unroll
            for (int j = 0; j < 4; ++j)
                acc[i][j] = wmma_bf16(a[i], b[j], acc[i][j]);
    }
    const int half = lane >> 4;
#pragma unroll
    for (int i = 0; i < 4; ++i)
#pragma unroll
        for (int j = 0; j < 4; ++j)
#pragma unroll
            for (int r = 0; r < 8; ++r)
                X[(size_t)(tm + i * 16 + r + 8 * half) * LDC + tn + j * 16 + r16] = acc[i][j][r];
}

// ---------------------------------------------------------------------------
// Kernel 3: per-half column BatchNorm stats (mean, rstd) over 4096 rows
// ---------------------------------------------------------------------------
__global__ void bnstats_kernel(const float* __restrict__ X, int cols,
                               float* __restrict__ mean, float* __restrict__ rstd) {
    int col = blockIdx.x * blockDim.x + threadIdx.x;
    if (col >= cols) return;
    int half = blockIdx.y;
    const float* p = X + (size_t)half * NBATCH * cols + col;
    float s = 0.f, ss = 0.f;
    for (int r = 0; r < NBATCH; ++r) {
        float v = p[(size_t)r * cols];
        s += v; ss += v * v;
    }
    float m = s * (1.f / NBATCH);
    float var = ss * (1.f / NBATCH) - m * m;
    mean[half * cols + col] = m;
    rstd[half * cols + col] = rsqrtf(var + BN_EPS);
}

// ---------------------------------------------------------------------------
// Kernel 4: BN1 apply + affine + ReLU + bf16 cast
// ---------------------------------------------------------------------------
__global__ void bnapply_kernel(const float* __restrict__ X,
                               const float* __restrict__ mean, const float* __restrict__ rstd,
                               const float* __restrict__ gamma, const float* __restrict__ beta,
                               __bf16* __restrict__ Xr) {
    size_t idx = (size_t)blockIdx.x * blockDim.x + threadIdx.x;
    if (idx >= (size_t)M2 * LATENT) return;
    int col  = (int)(idx & (LATENT - 1));
    int row  = (int)(idx >> 11);           // LATENT == 2048 == 1<<11
    int half = row >> 12;                  // 4096 rows per half
    float xn = (X[idx] - mean[half * LATENT + col]) * rstd[half * LATENT + col]
               * gamma[col] + beta[col];
    Xr[idx] = f2bf(fmaxf(xn, 0.f));
}

// ---------------------------------------------------------------------------
// Kernel 5: GEMM2  Y[8192,128] = Xr[8192,2048] @ W2[128,2048]^T
// Each wave: 16 rows x 128 cols (A fragment reused across 8 N-tiles).
// ---------------------------------------------------------------------------
__global__ __launch_bounds__(256) void gemm2_kernel(
        const __bf16* __restrict__ Xr, const __bf16* __restrict__ W2b,
        float* __restrict__ Y) {
    const int K = LATENT;
    const int lane = threadIdx.x & 31, wave = threadIdx.x >> 5;
    const int tm = blockIdx.x * 128 + wave * 16;
    const int r16 = lane & 15;

    const __bf16* ar = Xr + (size_t)(tm + r16) * K;
    const __bf16* br[8];
#pragma unroll
    for (int j = 0; j < 8; ++j) br[j] = W2b + (size_t)(j * 16 + r16) * K;

    v8f acc[8];
#pragma unroll
    for (int j = 0; j < 8; ++j) acc[j] = (v8f)0.f;

    for (int k = 0; k < K; k += 32) {
        v16bf a = load_frag(ar, k);
#pragma unroll
        for (int j = 0; j < 8; ++j)
            acc[j] = wmma_bf16(a, load_frag(br[j], k), acc[j]);
    }
    const int half = lane >> 4;
#pragma unroll
    for (int j = 0; j < 8; ++j)
#pragma unroll
        for (int r = 0; r < 8; ++r)
            Y[(size_t)(tm + r + 8 * half) * PROJ + j * 16 + r16] = acc[j][r];
}

// ---------------------------------------------------------------------------
// Kernel 6: BN2 (no affine) + row L2-normalize; emit ZA=[zn1;zn2], ZB=[zn2;zn1]
// ---------------------------------------------------------------------------
__global__ void norm_kernel(const float* __restrict__ Y,
                            const float* __restrict__ mean2, const float* __restrict__ rstd2,
                            __bf16* __restrict__ ZA, __bf16* __restrict__ ZB) {
    int row = blockIdx.x * blockDim.x + threadIdx.x;
    if (row >= M2) return;
    int half = row >> 12;
    const float* y  = Y + (size_t)row * PROJ;
    const float* m  = mean2 + half * PROJ;
    const float* rs = rstd2 + half * PROJ;
    float ss = 0.f;
    for (int c = 0; c < PROJ; ++c) {
        float z = (y[c] - m[c]) * rs[c];
        ss += z * z;
    }
    float inv = 1.f / fmaxf(sqrtf(ss), COS_EPS);
    int zbrow = row ^ NBATCH;      // swap halves
    for (int c = 0; c < PROJ; ++c) {
        float z = (y[c] - m[c]) * rs[c] * inv;
        __bf16 b = f2bf(z);
        ZA[(size_t)row * PROJ + c]   = b;
        ZB[(size_t)zbrow * PROJ + c] = b;
    }
}

// ---------------------------------------------------------------------------
// Kernel 7: sim GEMM  out[8192,8192] = ZA @ ZB^T * (1/T), |r-c|==4096 -> -inf
// ---------------------------------------------------------------------------
__global__ __launch_bounds__(256) void sim_kernel(
        const __bf16* __restrict__ ZA, const __bf16* __restrict__ ZB,
        float* __restrict__ out) {
    const int K = PROJ;
    const int lane = threadIdx.x & 31, wave = threadIdx.x >> 5;
    const int tm = blockIdx.y * 128 + (wave & 1) * 64;
    const int tn = blockIdx.x * 256 + (wave >> 1) * 64;
    const int r16 = lane & 15;

    const __bf16* ar[4]; const __bf16* br[4];
#pragma unroll
    for (int i = 0; i < 4; ++i) {
        ar[i] = ZA + (size_t)(tm + i * 16 + r16) * K;
        br[i] = ZB + (size_t)(tn + i * 16 + r16) * K;
    }
    v8f acc[4][4];
#pragma unroll
    for (int i = 0; i < 4; ++i)
#pragma unroll
        for (int j = 0; j < 4; ++j) acc[i][j] = (v8f)0.f;

#pragma unroll
    for (int k = 0; k < K; k += 32) {
        v16bf a[4], b[4];
#pragma unroll
        for (int i = 0; i < 4; ++i) a[i] = load_frag(ar[i], k);
#pragma unroll
        for (int j = 0; j < 4; ++j) b[j] = load_frag(br[j], k);
#pragma unroll
        for (int i = 0; i < 4; ++i)
#pragma unroll
            for (int j = 0; j < 4; ++j)
                acc[i][j] = wmma_bf16(a[i], b[j], acc[i][j]);
    }
    const int half = lane >> 4;
#pragma unroll
    for (int i = 0; i < 4; ++i)
#pragma unroll
        for (int j = 0; j < 4; ++j)
#pragma unroll
            for (int r = 0; r < 8; ++r) {
                int row = tm + i * 16 + r + 8 * half;
                int col = tn + j * 16 + r16;
                float v = INV_T * acc[i][j][r];
                if ((row - col == NBATCH) || (col - row == NBATCH)) v = -INFINITY;
                out[(size_t)row * M2 + col] = v;
            }
}

// ---------------------------------------------------------------------------
// Kernel 8: targets = arange(8192), int32 appended after the 8192^2 floats
// ---------------------------------------------------------------------------
__global__ void tgt_kernel(int* __restrict__ t) {
    int i = blockIdx.x * blockDim.x + threadIdx.x;
    if (i < M2) t[i] = i;
}

// ---------------------------------------------------------------------------
// Launch
// ---------------------------------------------------------------------------
extern "C" void kernel_launch(void* const* d_in, const int* in_sizes, int n_in,
                              void* d_out, int out_size, void* d_ws, size_t ws_size,
                              hipStream_t stream) {
    const float* h1     = (const float*)d_in[0];
    const float* h2     = (const float*)d_in[1];
    const float* W1     = (const float*)d_in[2];
    const float* gamma1 = (const float*)d_in[3];
    const float* beta1  = (const float*)d_in[4];
    const float* W2     = (const float*)d_in[5];

    char* ws = (char*)d_ws;
    // Workspace layout (bytes). Hbf region is reused for Xr after GEMM1+BN1.
    __bf16* Hbf  = (__bf16*)(ws);                         // [8192,2048] bf16 (32 MB)
    __bf16* Xr   = (__bf16*)(ws);                         //  reuse: BN1(ReLU) output
    size_t  o    = (size_t)M2 * LATENT * 2;
    __bf16* W1b  = (__bf16*)(ws + o);  o += (size_t)LATENT * LATENT * 2;   // 8 MB
    __bf16* W2b  = (__bf16*)(ws + o);  o += (size_t)PROJ * LATENT * 2;     // 0.5 MB
    float*  X    = (float*) (ws + o);  o += (size_t)M2 * LATENT * 4;       // 64 MB
    float*  Y    = (float*) (ws + o);  o += (size_t)M2 * PROJ * 4;         // 4 MB
    __bf16* ZA   = (__bf16*)(ws + o);  o += (size_t)M2 * PROJ * 2;         // 2 MB
    __bf16* ZB   = (__bf16*)(ws + o);  o += (size_t)M2 * PROJ * 2;         // 2 MB
    float*  mean1 = (float*)(ws + o);  o += 2 * LATENT * 4;
    float*  rstd1 = (float*)(ws + o);  o += 2 * LATENT * 4;
    float*  mean2 = (float*)(ws + o);  o += 2 * PROJ * 4;
    float*  rstd2 = (float*)(ws + o);  o += 2 * PROJ * 4;

    const int nh = NBATCH * LATENT;            // 8,388,608
    const int nw1 = LATENT * LATENT;           // 4,194,304
    const int nw2 = PROJ * LATENT;             // 262,144

    // 1) convert operands to bf16 (h1;h2 stacked)
    cvt_f32_bf16<<<nh / 256, 256, 0, stream>>>(h1, Hbf, nh);
    cvt_f32_bf16<<<nh / 256, 256, 0, stream>>>(h2, Hbf + (size_t)nh, nh);
    cvt_f32_bf16<<<nw1 / 256, 256, 0, stream>>>(W1, W1b, nw1);
    cvt_f32_bf16<<<nw2 / 256, 256, 0, stream>>>(W2, W2b, nw2);

    // 2) GEMM1: X = H @ W1^T
    gemm1_kernel<<<dim3(LATENT / 256, M2 / 128), 256, 0, stream>>>(Hbf, W1b, X);

    // 3) BN1 stats per half, then apply+ReLU -> bf16 (overwrites Hbf region)
    bnstats_kernel<<<dim3(LATENT / 256, 2), 256, 0, stream>>>(X, LATENT, mean1, rstd1);
    bnapply_kernel<<<(unsigned)(((size_t)M2 * LATENT) / 256), 256, 0, stream>>>(
        X, mean1, rstd1, gamma1, beta1, Xr);

    // 4) GEMM2: Y = Xr @ W2^T
    gemm2_kernel<<<M2 / 128, 256, 0, stream>>>(Xr, W2b, Y);

    // 5) BN2 stats (no affine) + row normalize -> ZA/ZB bf16
    bnstats_kernel<<<dim3(1, 2), 128, 0, stream>>>(Y, PROJ, mean2, rstd2);
    norm_kernel<<<M2 / 256, 256, 0, stream>>>(Y, mean2, rstd2, ZA, ZB);

    // 6) sim GEMM -> d_out (scale + diag mask fused), then targets
    float* out = (float*)d_out;
    sim_kernel<<<dim3(M2 / 256, M2 / 128), 256, 0, stream>>>(ZA, ZB, out);
    tgt_kernel<<<M2 / 256, 256, 0, stream>>>((int*)((float*)d_out + (size_t)M2 * M2));
}
